// GNNAttentionClassifier_64527588655726
// MI455X (gfx1250) — compile-verified, matrix operand order checked by reading, not run on hardware
//
#include <hip/hip_runtime.h>
#include <hip/hip_bf16.h>
#include <math.h>

typedef __attribute__((ext_vector_type(16))) __bf16 v16bf;
typedef __attribute__((ext_vector_type(8)))  float  v8f;

// ---------- helpers ----------
__device__ inline __bf16 f2bf(float f) {
    union { float f; unsigned u; } c; c.f = f;
    unsigned r = (c.u + 0x7FFFu + ((c.u >> 16) & 1u)) >> 16;
    unsigned short s = (unsigned short)r;
    union { unsigned short u; __bf16 b; } o; o.u = s;
    return o.b;
}
__device__ inline __bf16 bf_zero() {
    union { unsigned short u; __bf16 b; } o; o.u = 0;
    return o.b;
}
// Inverse of the 16-bit WMMA A/B fragment K-mapping (ISA 7.12.2).
// Given k within a 32-wide chunk (kk) and the 0..15 row/col index (lm),
// returns the bf16 slot (lane*16 + e) inside a 512-element fragment.
__device__ inline int frag_idx(int kk, int lm) {
    int lane = lm + (((kk >> 3) & 1) << 4);
    int e    = (kk & 7) + (((kk >> 4) & 1) << 3);
    return lane * 16 + e;
}

// ---------- Stage 1: attention pooling over each node's set ----------
#define PS 20   // set size
#define PD 128  // feature dim
#define PH 64   // embed hidden
#define PO 128  // pooled out

__global__ void __launch_bounds__(128)
pool_kernel(const float* __restrict__ feats,
            const float* __restrict__ W_embed, const float* __restrict__ b_embed,
            const float* __restrict__ W_attn,  const float* __restrict__ b_attn,
            const float* __restrict__ W_out,   const float* __restrict__ b_out,
            float* __restrict__ pooled, int N)
{
    // W_embed in bf16 WMMA fragment order: frag = kc*4+nt (kc<4, nt<4), 512 elems each
    __shared__ __align__(32) __bf16 sWe[16 * 512];          // 16 KB
    __shared__ float  sWo[PH * PO];                          // 32 KB
    __shared__ float  sWa[PH];
    __shared__ float  sBe[PH];
    __shared__ float  sBo[PO];
    // per-wave node tile in fragment order: frag = mt*4+kc (mt<2, kc<4)
    __shared__ __align__(32) __bf16 sXf[4][8 * 512];         // 32 KB
    __shared__ float  sZ[4][PS * PH];                        // 20 KB
    __shared__ float  sAttn[4][PS];
    __shared__ float  sPH[4][PH];

    const int tid = threadIdx.x;
    // stage W_embed (PD x PH) into fragment order
    for (int i = tid; i < PD * PH; i += 128) {
        int kg = i >> 6, n = i & 63;                 // row k, col n
        int kc = kg >> 5, kk = kg & 31, nt = n >> 4, lm = n & 15;
        sWe[(kc * 4 + nt) * 512 + frag_idx(kk, lm)] = f2bf(W_embed[i]);
    }
    for (int i = tid; i < PH * PO; i += 128) sWo[i] = W_out[i];
    if (tid < PH) { sWa[tid] = W_attn[tid]; sBe[tid] = b_embed[tid]; }
    if (tid < PO) sBo[tid] = b_out[tid];

    const int wave = tid >> 5, lane = tid & 31;
    const int node = blockIdx.x * 4 + wave;
    if (node < N) {
        // zero-pad fragments for rows 16..31 (mt==1), then scatter the 20 rows
        unsigned* xz = (unsigned*)(sXf[wave] + 4 * 512);
        for (int i = lane; i < 1024; i += 32) xz[i] = 0u;
        const float* xp = feats + (size_t)node * (PS * PD);
        for (int i = lane; i < PS * PD; i += 32) {
            int s = i >> 7, k = i & 127;
            int mt = s >> 4, lm = s & 15, kc = k >> 5, kk = k & 31;
            sXf[wave][(mt * 4 + kc) * 512 + frag_idx(kk, lm)] = f2bf(xp[i]);
        }
    }
    __syncthreads();
    if (node >= N) return;   // wave-uniform

    const int ml  = lane & 15;
    const int hi8 = (lane & 16) ? 8 : 0;
    const v16bf* xf  = (const v16bf*)sXf[wave];
    const v16bf* wef = (const v16bf*)sWe;

    // z = relu(X @ W_embed + b_embed); 32 WMMAs, each fragment read once (2x b128)
    v8f vacc[4][2];
#pragma unroll
    for (int nt = 0; nt < 4; ++nt) { vacc[nt][0] = (v8f){}; vacc[nt][1] = (v8f){}; }
#pragma unroll
    for (int kc = 0; kc < 4; ++kc) {
        v16bf a0 = xf[(0 * 4 + kc) * 32 + lane];
        v16bf a1 = xf[(1 * 4 + kc) * 32 + lane];
#pragma unroll
        for (int nt = 0; nt < 4; ++nt) {
            v16bf b = wef[(kc * 4 + nt) * 32 + lane];
            vacc[nt][0] = __builtin_amdgcn_wmma_f32_16x16x32_bf16(false, a0, false, b, (short)0, vacc[nt][0], false, false);
            vacc[nt][1] = __builtin_amdgcn_wmma_f32_16x16x32_bf16(false, a1, false, b, (short)0, vacc[nt][1], false, false);
        }
    }
#pragma unroll
    for (int nt = 0; nt < 4; ++nt) {
        int ncol = nt * 16 + ml;
        float bias = sBe[ncol];
#pragma unroll
        for (int r = 0; r < 8; ++r) {
            int row0 = r + hi8;                       // 0..15, always valid
            float v0 = vacc[nt][0][r] + bias; v0 = v0 > 0.f ? v0 : 0.f;
            sZ[wave][row0 * PH + ncol] = v0;
            int row1 = 16 + r + hi8;                  // 16..31, valid < 20
            if (row1 < PS) {
                float v1 = vacc[nt][1][r] + bias; v1 = v1 > 0.f ? v1 : 0.f;
                sZ[wave][row1 * PH + ncol] = v1;
            }
        }
    }

    // scores + softmax over the 20 set elements (lane s handles row s)
    float score = -INFINITY;
    if (lane < PS) {
        float a = b_attn[0];
        for (int h = 0; h < PH; ++h) a += sZ[wave][lane * PH + h] * sWa[h];
        score = a;
    }
    float m = score;
    for (int off = 16; off > 0; off >>= 1) m = fmaxf(m, __shfl_xor(m, off, 32));
    float ev = (lane < PS) ? __expf(score - m) : 0.f;
    float ssum = ev;
    for (int off = 16; off > 0; off >>= 1) ssum += __shfl_xor(ssum, off, 32);
    if (lane < PS) sAttn[wave][lane] = ev / ssum;

    // pooled_h[h] = sum_s attn[s] * z[s,h]   (each lane: 2 columns)
    float p0 = 0.f, p1 = 0.f;
    for (int s = 0; s < PS; ++s) {
        float a = sAttn[wave][s];
        p0 += a * sZ[wave][s * PH + lane];
        p1 += a * sZ[wave][s * PH + lane + 32];
    }
    sPH[wave][lane] = p0; sPH[wave][lane + 32] = p1;

    // pooled = pooled_h @ W_out + b_out   (each lane: 4 columns)
    for (int j = 0; j < 4; ++j) {
        int c = lane + 32 * j;
        float acc = sBo[c];
        for (int h = 0; h < PH; ++h) acc += sPH[wave][h] * sWo[h * PO + c];
        pooled[(size_t)node * PO + c] = acc;
    }
}

// ---------- generic tiled bf16 WMMA GEMM:  C = A(MxK) @ W(KxNCOLS), K<=128 ----------
template <int NCOLS>
__global__ void __launch_bounds__(256)
gemm_bf16_kernel(const float* __restrict__ A, const float* __restrict__ W,
                 float* __restrict__ C, int M, int K)
{
    constexpr int NT = NCOLS / 16;
    __shared__ __align__(32) __bf16 sW[(128 / 32) * NT * 512];   // fragment order
    const int tid = threadIdx.x;
    for (int i = tid; i < K * NCOLS; i += 256) {
        int kg = i / NCOLS, n = i - kg * NCOLS;
        int kc = kg >> 5, kk = kg & 31, nt = n >> 4, lm = n & 15;
        sW[(kc * NT + nt) * 512 + frag_idx(kk, lm)] = f2bf(W[i]);
    }
    __syncthreads();

    const int wave = tid >> 5, lane = tid & 31;
    const int mtile = blockIdx.x * 8 + wave;
    const int row_base = mtile * 16;
    if (row_base >= M) return;      // wave-uniform

    const int ml  = lane & 15;
    const int hi8 = (lane & 16) ? 8 : 0;
    const int arow = row_base + ml;
    const bool arv = arow < M;
    const __bf16 zb = bf_zero();
    const v16bf* wf = (const v16bf*)sW;

    v8f acc[NT];
#pragma unroll
    for (int nt = 0; nt < NT; ++nt) acc[nt] = (v8f){};

    const int kcs = K >> 5;
    for (int kc = 0; kc < kcs; ++kc) {
        v16bf a;
        if (arv) {
            // two contiguous 8-float runs per lane -> 4x global_load_b128
            const float4* ap = (const float4*)(A + (size_t)arow * K + kc * 32 + (hi8 ? 8 : 0));
            float4 f0 = ap[0], f1 = ap[1], f2 = ap[4], f3 = ap[5];
            a[0]  = f2bf(f0.x); a[1]  = f2bf(f0.y); a[2]  = f2bf(f0.z); a[3]  = f2bf(f0.w);
            a[4]  = f2bf(f1.x); a[5]  = f2bf(f1.y); a[6]  = f2bf(f1.z); a[7]  = f2bf(f1.w);
            a[8]  = f2bf(f2.x); a[9]  = f2bf(f2.y); a[10] = f2bf(f2.z); a[11] = f2bf(f2.w);
            a[12] = f2bf(f3.x); a[13] = f2bf(f3.y); a[14] = f2bf(f3.z); a[15] = f2bf(f3.w);
        } else {
#pragma unroll
            for (int e = 0; e < 16; ++e) a[e] = zb;
        }
#pragma unroll
        for (int nt = 0; nt < NT; ++nt) {
            v16bf b = wf[(kc * NT + nt) * 32 + lane];
            acc[nt] = __builtin_amdgcn_wmma_f32_16x16x32_bf16(false, a, false, b, (short)0, acc[nt], false, false);
        }
    }
#pragma unroll
    for (int nt = 0; nt < NT; ++nt)
#pragma unroll
        for (int r = 0; r < 8; ++r) {
            int row = row_base + r + hi8;
            if (row < M) C[(size_t)row * NCOLS + nt * 16 + ml] = acc[nt][r];
        }
}

// ---------- degree / normalization ----------
__global__ void deg_init_kernel(float* deg, int N) {
    int i = blockIdx.x * blockDim.x + threadIdx.x;
    if (i < N) deg[i] = 1.0f;     // self-loop
}
__global__ void deg_scatter_kernel(const int* __restrict__ dst, int E, float* deg) {
    for (int i = blockIdx.x * blockDim.x + threadIdx.x; i < E; i += gridDim.x * blockDim.x)
        atomicAdd(&deg[dst[i]], 1.0f);
}
__global__ void deg_rsqrt_kernel(float* deg, int N) {
    int i = blockIdx.x * blockDim.x + threadIdx.x;
    if (i < N) deg[i] = rsqrtf(deg[i]);
}

// ---------- GCN aggregation ----------
template <int F>
__global__ void gcn_self_init_kernel(const float* __restrict__ h, const float* __restrict__ dinv,
                                     float* __restrict__ out, int N)
{
    size_t i = (size_t)blockIdx.x * blockDim.x + threadIdx.x;
    size_t tot = (size_t)N * F;
    if (i < tot) {
        int n = (int)(i / F);
        float dv = dinv[n];
        out[i] = h[i] * dv * dv;
    }
}
template <int F>
__global__ void gcn_scatter_kernel(const int* __restrict__ src, const int* __restrict__ dst, int E,
                                   const float* __restrict__ h, const float* __restrict__ dinv,
                                   float* __restrict__ out)
{
    size_t tot = (size_t)E * F;
    for (size_t i = (size_t)blockIdx.x * blockDim.x + threadIdx.x; i < tot;
         i += (size_t)gridDim.x * blockDim.x) {
        size_t e = i / F;
        int f = (int)(i - e * F);
        int s = src[e], d = dst[e];
        float c = dinv[s] * dinv[d];
        atomicAdd(&out[(size_t)d * F + f], h[(size_t)s * F + f] * c);
    }
}
template <int F>
__global__ void bias_relu_kernel(float* __restrict__ x, const float* __restrict__ b, int N)
{
    size_t i = (size_t)blockIdx.x * blockDim.x + threadIdx.x;
    size_t tot = (size_t)N * F;
    if (i < tot) {
        float v = x[i] + b[i % F];
        x[i] = v > 0.f ? v : 0.f;
    }
}

// ---------- MLP head on gathered queries ----------
__global__ void __launch_bounds__(256)
mlp_head_kernel(const float* __restrict__ x, const int* __restrict__ qidx,
                const float* __restrict__ Wm1, const float* __restrict__ bm1,
                const float* __restrict__ Wm2, const float* __restrict__ bm2,
                float* __restrict__ out, int B)
{
    __shared__ float sW1[32 * 64], sB1[64], sW2[64], sB2[1];
    int tid = threadIdx.x;
    for (int i = tid; i < 32 * 64; i += 256) sW1[i] = Wm1[i];
    if (tid < 64) { sB1[tid] = bm1[tid]; sW2[tid] = Wm2[tid]; }
    if (tid == 0) sB2[0] = bm2[0];
    __syncthreads();

    int q = blockIdx.x * blockDim.x + tid;
    if (q >= B) return;
    int n = qidx[q];
    float xi[32];
    const float* xp = x + (size_t)n * 32;
#pragma unroll
    for (int i = 0; i < 32; ++i) xi[i] = xp[i];
    float o = sB2[0];
    for (int j = 0; j < 64; ++j) {
        float hj = sB1[j];
#pragma unroll
        for (int i = 0; i < 32; ++i) hj += xi[i] * sW1[i * 64 + j];
        o += (hj > 0.f ? hj : 0.f) * sW2[j];
    }
    out[q] = o;
}

// ---------- driver ----------
extern "C" void kernel_launch(void* const* d_in, const int* in_sizes, int n_in,
                              void* d_out, int out_size, void* d_ws, size_t ws_size,
                              hipStream_t stream)
{
    const float* feats   = (const float*)d_in[0];
    const int*   ei      = (const int*)  d_in[2];
    const int*   qidx    = (const int*)  d_in[3];
    const float* W_embed = (const float*)d_in[4];
    const float* b_embed = (const float*)d_in[5];
    const float* W_attn  = (const float*)d_in[6];
    const float* b_attn  = (const float*)d_in[7];
    const float* W_out   = (const float*)d_in[8];
    const float* b_out   = (const float*)d_in[9];
    const float* W1      = (const float*)d_in[10];
    const float* b1      = (const float*)d_in[11];
    const float* W2      = (const float*)d_in[12];
    const float* b2      = (const float*)d_in[13];
    const float* Wm1     = (const float*)d_in[14];
    const float* bm1     = (const float*)d_in[15];
    const float* Wm2     = (const float*)d_in[16];
    const float* bm2     = (const float*)d_in[17];

    const int N = in_sizes[0] / (PS * PD);
    const int E = in_sizes[2] / 2;
    const int B = in_sizes[3];
    const int* src = ei;
    const int* dst = ei + E;

    // workspace layout (256B aligned)
    size_t off = 0;
    auto take = [&](size_t bytes) {
        size_t o = off; off += (bytes + 255) & ~(size_t)255; return o;
    };
    char* ws = (char*)d_ws;
    float* bufA = (float*)(ws + take((size_t)N * 128 * sizeof(float))); // pooled / h2
    float* bufB = (float*)(ws + take((size_t)N * 128 * sizeof(float))); // h1 / agg2 / x2
    float* bufC = (float*)(ws + take((size_t)N * 128 * sizeof(float))); // agg1 / x1
    float* dinv = (float*)(ws + take((size_t)N * sizeof(float)));

    // degree -> rsqrt
    deg_init_kernel<<<(N + 255) / 256, 256, 0, stream>>>(dinv, N);
    deg_scatter_kernel<<<2048, 256, 0, stream>>>(dst, E, dinv);
    deg_rsqrt_kernel<<<(N + 255) / 256, 256, 0, stream>>>(dinv, N);

    // attention pooling -> bufA (N x 128)
    pool_kernel<<<(N + 3) / 4, 128, 0, stream>>>(
        feats, W_embed, b_embed, W_attn, b_attn, W_out, b_out, bufA, N);

    // GCN layer 1: h1 = pooled @ W1 (N x 128)
    {
        int mtiles = (N + 15) / 16;
        gemm_bf16_kernel<128><<<(mtiles + 7) / 8, 256, 0, stream>>>(bufA, W1, bufB, N, 128);
    }
    {
        size_t tot = (size_t)N * 128;
        gcn_self_init_kernel<128><<<(int)((tot + 255) / 256), 256, 0, stream>>>(bufB, dinv, bufC, N);
        gcn_scatter_kernel<128><<<4096, 256, 0, stream>>>(src, dst, E, bufB, dinv, bufC);
        bias_relu_kernel<128><<<(int)((tot + 255) / 256), 256, 0, stream>>>(bufC, b1, N);
    }

    // GCN layer 2: h2 = x1 @ W2 (N x 32)
    {
        int mtiles = (N + 15) / 16;
        gemm_bf16_kernel<32><<<(mtiles + 7) / 8, 256, 0, stream>>>(bufC, W2, bufA, N, 128);
    }
    {
        size_t tot = (size_t)N * 32;
        gcn_self_init_kernel<32><<<(int)((tot + 255) / 256), 256, 0, stream>>>(bufA, dinv, bufB, N);
        gcn_scatter_kernel<32><<<4096, 256, 0, stream>>>(src, dst, E, bufA, dinv, bufB);
        bias_relu_kernel<32><<<(int)((tot + 255) / 256), 256, 0, stream>>>(bufB, b2, N);
    }

    // MLP head -> d_out (B logits)
    mlp_head_kernel<<<(B + 255) / 256, 256, 0, stream>>>(
        bufB, qidx, Wm1, bm1, Wm2, bm2, (float*)d_out, B);
}
